// Attention_45208825757848
// MI455X (gfx1250) — compile-verified, matrix operand order checked by reading, not run on hardware
//
#include <hip/hip_runtime.h>

typedef _Float16 half_t;
typedef __attribute__((ext_vector_type(16))) _Float16 v16h;
typedef __attribute__((ext_vector_type(8)))  _Float16 v8h;
typedef __attribute__((ext_vector_type(8)))  float    v8f;
typedef int v4i_vec __attribute__((vector_size(16)));

#define WMMA_F16(a, b, c) \
  __builtin_amdgcn_wmma_f32_16x16x32_f16(false, (a), false, (b), (short)0, (c), false, false)

// gfx1250 async global->LDS copy (ASYNCcnt path), guarded so a missing builtin
// degrades to a register-bounce copy instead of breaking the build.
#if defined(__gfx1250__) && __has_builtin(__builtin_amdgcn_global_load_async_to_lds_b128)
#define HAVE_ASYNC_LDS 1
#else
#define HAVE_ASYNC_LDS 0
#endif

#if HAVE_ASYNC_LDS
__device__ __forceinline__ void async_copy16(half_t* lds_dst, const half_t* gsrc) {
  __builtin_amdgcn_global_load_async_to_lds_b128(
      (__attribute__((address_space(1))) v4i_vec*)gsrc,
      (__attribute__((address_space(3))) v4i_vec*)lds_dst, 0, 0);
}
#endif

// ---------------------------------------------------------------------------
// Fragment loader for 16-bit A/B operands (CDNA5 WMMA layout):
// lane L holds row/col (L&15); VGPR0-3 = K k0..k0+7 (k0 = (L>>4)*8),
// VGPR4-7 = K k0+16..k0+23.  Two 16-byte loads per lane.
// ---------------------------------------------------------------------------
__device__ __forceinline__ v16h load_frag(const half_t* __restrict__ p, int ld) {
  int lane = threadIdx.x & 31;
  const half_t* q = p + (size_t)(lane & 15) * ld + ((lane >> 4) << 3);
  union { v16h v; v8h h2[2]; } u;
  u.h2[0] = *(const v8h*)q;
  u.h2[1] = *(const v8h*)(q + 16);
  return u.v;
}

// Same layout, reading a staged LDS panel with row stride 32 halves.
__device__ __forceinline__ v16h load_frag_lds(const half_t* p) {
  int lane = threadIdx.x & 31;
  const half_t* q = p + (lane & 15) * 32 + ((lane >> 4) << 3);
  union { v16h v; v8h h2[2]; } u;
  u.h2[0] = *(const v8h*)q;
  u.h2[1] = *(const v8h*)(q + 16);
  return u.v;
}

// ---------------------------------------------------------------------------
// Cast helpers
// ---------------------------------------------------------------------------
__global__ void cast_f32_to_f16(const float* __restrict__ src, half_t* __restrict__ dst, int n) {
  int i = blockIdx.x * 256 + threadIdx.x;
  if (i < n) dst[i] = (half_t)src[i];
}

// Wt[n][k] = W[k][n]  (1024x1024)
__global__ void transpose_cast_w(const float* __restrict__ W, half_t* __restrict__ Wt) {
  int idx = blockIdx.x * 256 + threadIdx.x;   // 1M threads
  int n = idx >> 10, k = idx & 1023;
  Wt[idx] = (half_t)W[(size_t)k * 1024 + n];
}

// ---------------------------------------------------------------------------
// RoPE on Q and K in place.  Layout [B,S,H*HD] f16, pair p of head h.
// ---------------------------------------------------------------------------
__global__ void rope_qk(half_t* __restrict__ Q, half_t* __restrict__ K,
                        const float* __restrict__ fcos, const float* __restrict__ fsin) {
  int idx = blockIdx.x * 256 + threadIdx.x;           // B*S*H*32 = 4194304
  int p = idx & 31;
  int h = (idx >> 5) & 15;
  int s = (idx >> 9) & 1023;
  int b = idx >> 19;
  float c  = fcos[s * 32 + p];
  float sn = fsin[s * 32 + p];
  size_t base = ((size_t)(b * 1024 + s)) * 1024 + h * 64 + 2 * p;
  float tr = (float)Q[base], ti = (float)Q[base + 1];
  Q[base]     = (half_t)(tr * c - ti * sn);
  Q[base + 1] = (half_t)(tr * sn + ti * c);
  tr = (float)K[base]; ti = (float)K[base + 1];
  K[base]     = (half_t)(tr * c - ti * sn);
  K[base + 1] = (half_t)(tr * sn + ti * c);
}

// ---------------------------------------------------------------------------
// Per-tile store epilogue.
// MODE 0: f16 row-major.  MODE 1: V-transposed [B][H][HD][S] f16.  MODE 2: f32.
// ---------------------------------------------------------------------------
template <int MODE>
__device__ __forceinline__ void store_tile(void* __restrict__ Cout, int N,
                                           int row0, int col0, const v8f& acc) {
  int lane = threadIdx.x & 31;
  int hi = lane >> 4, cl = lane & 15;
  if (MODE == 0) {
    half_t* C = (half_t*)Cout;
#pragma unroll
    for (int v = 0; v < 8; ++v)
      C[(size_t)(row0 + v + 8 * hi) * N + col0 + cl] = (half_t)acc[v];
  } else if (MODE == 1) {
    half_t* Vt = (half_t*)Cout;   // [B][H][HD][S]
#pragma unroll
    for (int v = 0; v < 8; ++v) {
      int r = row0 + v + 8 * hi;
      int b = r >> 10, s = r & 1023;
      int c = col0 + cl;
      Vt[(((size_t)b * 16 + (c >> 6)) * 64 + (c & 63)) * 1024 + s] = (half_t)acc[v];
    }
  } else {
    float* C = (float*)Cout;
#pragma unroll
    for (int v = 0; v < 8; ++v)
      C[(size_t)(row0 + v + 8 * hi) * N + col0 + cl] = acc[v];
  }
}

// ---------------------------------------------------------------------------
// WMMA GEMM: C[M,N] = A[M,K] * Bt[N,K]^T.  256 threads = 8 waves; each wave
// computes a 32x32 tile (4 WMMAs / k-step); block covers 128x64.
// The 64x32 B panel is staged cooperatively in LDS (async global->LDS when
// available), reused by all 4 row-waves.  A fragments come straight from
// global (hot in the 192MB L2), with global_prefetch for the next k-step.
// ---------------------------------------------------------------------------
template <int MODE>
__global__ void gemm_wmma(const half_t* __restrict__ A, const half_t* __restrict__ Bt,
                          void* __restrict__ Cout, int M, int N, int K) {
  __shared__ __align__(16) half_t bsh[64][32];   // [block col][k] panel, 4KB
  int tid  = threadIdx.x;
  int wave = tid >> 5;
  int lane = tid & 31;
  int row0 = blockIdx.y * 128 + (wave >> 1) * 32;
  int col0 = blockIdx.x * 64 + (wave & 1) * 32;
  int colblk = blockIdx.x * 64;

  v8f acc00 = {}, acc01 = {}, acc10 = {}, acc11 = {};
  const half_t* A0 = A + (size_t)row0 * K;
  const half_t* A1 = A + (size_t)(row0 + 16) * K;

  // Staging assignment: thread copies 16B chunk (sn, skc..skc+7) of the panel.
  int sn  = tid >> 2;
  int skc = (tid & 3) * 8;
  const half_t* gB = Bt + (size_t)(colblk + sn) * K + skc;
  int cLocal = (wave & 1) * 32;

  for (int k = 0; k < K; k += 32) {
#if HAVE_ASYNC_LDS
    async_copy16(&bsh[sn][skc], gB + k);
#else
    *(v8h*)&bsh[sn][skc] = *(const v8h*)(gB + k);
#endif
    v16h a0 = load_frag(A0 + k, K);
    v16h a1 = load_frag(A1 + k, K);
    if (k + 32 < K) {
      __builtin_prefetch(A0 + (size_t)(lane & 15) * K + k + 32, 0, 0);
      __builtin_prefetch(A1 + (size_t)(lane & 15) * K + k + 32, 0, 0);
    }
#if HAVE_ASYNC_LDS
    asm volatile("s_wait_asynccnt 0x0" ::: "memory");
#endif
    __syncthreads();
    v16h b0 = load_frag_lds(&bsh[cLocal][0]);
    v16h b1 = load_frag_lds(&bsh[cLocal + 16][0]);
    acc00 = WMMA_F16(a0, b0, acc00);
    acc01 = WMMA_F16(a0, b1, acc01);
    acc10 = WMMA_F16(a1, b0, acc10);
    acc11 = WMMA_F16(a1, b1, acc11);
    __syncthreads();   // panel consumed before next overwrite
  }

  store_tile<MODE>(Cout, N, row0,      col0,      acc00);
  store_tile<MODE>(Cout, N, row0,      col0 + 16, acc01);
  store_tile<MODE>(Cout, N, row0 + 16, col0,      acc10);
  store_tile<MODE>(Cout, N, row0 + 16, col0 + 16, acc11);
}

// ---------------------------------------------------------------------------
// Flash attention, block-causal (BS=8).  One wave per (b, h, 16-row q-tile).
// blockDim = 256 (8 waves), grid = B*H*(S/16)/8 = 1024 blocks.
// ---------------------------------------------------------------------------
__global__ void attn_flash(const half_t* __restrict__ Q, const half_t* __restrict__ Kb,
                           const half_t* __restrict__ Vt, half_t* __restrict__ O) {
  __shared__ half_t psh[8][16 * 17];
  int wv   = threadIdx.x >> 5;
  int lane = threadIdx.x & 31;
  int w  = blockIdx.x * 8 + wv;
  int qt = w & 63;
  int h  = (w >> 6) & 15;
  int b  = w >> 10;
  int qb = qt * 16;
  int hi = lane >> 4, cl = lane & 15;
  int k0 = hi << 3;

  const size_t qbase = ((size_t)b * 1024 + qb) * 1024 + h * 64;
  v16h qa0 = load_frag(Q + qbase,      1024);   // head dims 0..31
  v16h qa1 = load_frag(Q + qbase + 32, 1024);   // head dims 32..63

  v8f o[4] = {};
  float rm[8], rl[8];
#pragma unroll
  for (int v = 0; v < 8; ++v) { rm[v] = -3.0e38f; rl[v] = 0.0f; }

  const half_t* vbase = Vt + (size_t)(b * 16 + h) * 64 * 1024;  // [HD][S]

  for (int kt = 0; kt <= qt; ++kt) {
    const half_t* kp = Kb + ((size_t)b * 1024 + kt * 16) * 1024 + h * 64;
    if (kt < qt)   // prefetch next key tile row for this lane
      __builtin_prefetch(kp + 16 * 1024 + (size_t)(lane & 15) * 1024, 0, 0);
    v16h kb0 = load_frag(kp,      1024);
    v16h kb1 = load_frag(kp + 32, 1024);
    v8f s = {};
    s = WMMA_F16(qa0, kb0, s);
    s = WMMA_F16(qa1, kb1, s);

    // Diagonal tile: keys j=8..15 (lanes 8..15, group 0) are disallowed for
    // q rows 0..7 (all VGPRs of those lanes).
    bool maskme = (kt == qt) && (hi == 0) && (cl >= 8);

#pragma unroll
    for (int v = 0; v < 8; ++v) {
      float x = (float)s[v] * 0.125f;           // 1/sqrt(64)
      if (maskme) x = -3.0e38f;
      float mx = x;
#pragma unroll
      for (int off = 1; off < 16; off <<= 1) mx = fmaxf(mx, __shfl_xor(mx, off));
      float mnew  = fmaxf(rm[v], mx);
      float alpha = __expf(rm[v] - mnew);
      float p     = __expf(x - mnew);
      float rs = p;
#pragma unroll
      for (int off = 1; off < 16; off <<= 1) rs += __shfl_xor(rs, off);
      rl[v] = rl[v] * alpha + rs;
      rm[v] = mnew;
      o[0][v] *= alpha; o[1][v] *= alpha; o[2][v] *= alpha; o[3][v] *= alpha;
      psh[wv][(v + 8 * hi) * 17 + cl] = (half_t)p;   // P row-major 16x16
    }

    // Reload P as A operand, zero-padded to K=32 (elements 8..15 = K 16..31).
    v16h pv;
#pragma unroll
    for (int j = 0; j < 8; ++j) pv[j] = psh[wv][cl * 17 + k0 + j];
#pragma unroll
    for (int j = 8; j < 16; ++j) pv[j] = (half_t)0.0f;

#pragma unroll
    for (int n = 0; n < 4; ++n) {
      const half_t* vr = vbase + (size_t)(n * 16 + cl) * 1024 + kt * 16 + k0;
      union { v16h v; v8h h2[2]; } vu;
      vu.h2[0] = *(const v8h*)vr;   // keys kt*16+k0 .. +7 for column d
      vu.h2[1] = vu.h2[0];          // K 16..31 multiplied by zero P — don't care
      o[n] = WMMA_F16(pv, vu.v, o[n]);
    }
  }

#pragma unroll
  for (int n = 0; n < 4; ++n)
#pragma unroll
    for (int v = 0; v < 8; ++v) {
      int r = qb + v + 8 * hi;
      O[((size_t)b * 1024 + r) * 1024 + h * 64 + n * 16 + cl] =
          (half_t)(o[n][v] / rl[v]);
    }
}

// ---------------------------------------------------------------------------
// Launch
// ---------------------------------------------------------------------------
extern "C" void kernel_launch(void* const* d_in, const int* in_sizes, int n_in,
                              void* d_out, int out_size, void* d_ws, size_t ws_size,
                              hipStream_t stream) {
  (void)in_sizes; (void)n_in; (void)out_size; (void)ws_size;
  const float* x    = (const float*)d_in[0];
  const float* wq   = (const float*)d_in[1];
  const float* wk   = (const float*)d_in[2];
  const float* wvp  = (const float*)d_in[3];
  const float* wo   = (const float*)d_in[4];
  const float* fcos = (const float*)d_in[5];
  const float* fsin = (const float*)d_in[6];

  const int M = 8192, N = 1024, K = 1024;
  half_t* ws  = (half_t*)d_ws;
  half_t* xb  = ws;                       // 8192*1024
  half_t* wqT = ws + 8388608;             // 1024*1024 each
  half_t* wkT = wqT + 1048576;
  half_t* wvT = wkT + 1048576;
  half_t* woT = wvT + 1048576;
  half_t* Qb  = ws + 12582912;            // 8192*1024
  half_t* Kbf = Qb + 8388608;             // 8192*1024
  half_t* Vt  = Kbf + 8388608;            // [B][H][HD][S]
  half_t* Ob  = xb;                       // reuse: xb dead after V projection

  cast_f32_to_f16<<<(M * K) / 256, 256, 0, stream>>>(x, xb, M * K);
  transpose_cast_w<<<(K * N) / 256, 256, 0, stream>>>(wq, wqT);
  transpose_cast_w<<<(K * N) / 256, 256, 0, stream>>>(wk, wkT);
  transpose_cast_w<<<(K * N) / 256, 256, 0, stream>>>(wvp, wvT);
  transpose_cast_w<<<(K * N) / 256, 256, 0, stream>>>(wo, woT);

  dim3 ggrid(N / 64, M / 128);
  gemm_wmma<0><<<ggrid, 256, 0, stream>>>(xb, wqT, Qb,  M, N, K);
  gemm_wmma<0><<<ggrid, 256, 0, stream>>>(xb, wkT, Kbf, M, N, K);
  gemm_wmma<1><<<ggrid, 256, 0, stream>>>(xb, wvT, Vt,  M, N, K);

  rope_qk<<<(8 * 1024 * 16 * 32) / 256, 256, 0, stream>>>(Qb, Kbf, fcos, fsin);

  attn_flash<<<1024, 256, 0, stream>>>(Qb, Kbf, Vt, Ob);

  gemm_wmma<2><<<ggrid, 256, 0, stream>>>(Ob, woT, d_out, M, N, K);
}
